// UNet_13185549598898
// MI455X (gfx1250) — matrix operand order repaired and checked by reading, not compiled, not run on hardware
//
#include <hip/hip_runtime.h>
#include <hip/hip_bf16.h>
#include <math.h>

typedef __attribute__((ext_vector_type(16))) _Float16 v16h;
typedef __attribute__((ext_vector_type(8)))  _Float16 h8;
typedef __attribute__((ext_vector_type(4)))  _Float16 h4;
typedef __attribute__((ext_vector_type(8)))  float    v8f;
typedef __attribute__((ext_vector_type(4)))  float    f4;

#define HEADS 8
#define EPSBN 1e-5f

// ---------------------------------------------------------------------------
// WMMA GEMM: C[M,N] = act( A[M,K] * B[K,N] (+bias) (+residual) )
// A,B fp32 in global, converted to f16 while staging into LDS.
// Requires K % 32 == 0 and N % 16 == 0 (guaranteed by caller).
// Block = 128 threads (4 waves); block computes a 64x16 C tile, one 16x16
// WMMA tile per wave. Double-buffered LDS staging with register-held global
// tiles: next tile's global_load_b128s are issued before the current WMMA,
// one barrier per K-step. Out-of-range rows handled by clamped (always
// valid) addresses + zeroing, so all loads are unconditional and wide.
// act: 0=none, 1=relu, 2=elu
// ---------------------------------------------------------------------------
__global__ __launch_bounds__(128)
void gemm_wmma_kernel(const float* __restrict__ A, const float* __restrict__ Bw,
                      float* __restrict__ C, int M, int N, int K,
                      const float* __restrict__ bias, const float* __restrict__ res,
                      int act)
{
    __shared__ alignas(16) _Float16 sA[2][64 * 40];   // 64 rows x 32 K (pad 40)
    __shared__ alignas(16) _Float16 sBt[2][16 * 40];  // 16 cols x 32 K, transposed

    const int tid  = threadIdx.x;
    const int wave = tid >> 5;
    const int lane = tid & 31;
    const int m0   = blockIdx.y * 64;
    const int n0   = blockIdx.x * 16;
    const int row  = lane & 15;   // M-row within tile (A) / N-col (B,C)
    const int hv   = lane >> 4;   // lane-group select

    // A staging: 64x32 floats = 512 float4; thread owns 4 (lin = tid + i*128)
    // B staging: 32x16 floats = 128 float4; thread owns 1
    const int rB  = tid >> 2;         // K-row 0..31
    const int cB4 = tid & 3;          // col group 0..3
    const f4 fzero = {0.0f, 0.0f, 0.0f, 0.0f};

    f4 ra[4];
    f4 rb;

    auto loadTiles = [&](int k0) {
        #pragma unroll
        for (int i = 0; i < 4; ++i) {
            int lin = tid + i * 128;
            int r = lin >> 3, c4 = lin & 7;
            int gm  = m0 + r;
            int gmc = (gm < M) ? gm : (M - 1);           // always-valid address
            f4 v = *(const f4*)(A + (size_t)gmc * K + k0 + c4 * 4);
            if (gm >= M) v = fzero;
            ra[i] = v;
        }
        rb = *(const f4*)(Bw + (size_t)(k0 + rB) * N + n0 + cB4 * 4);
    };
    auto storeTiles = [&](int buf) {
        #pragma unroll
        for (int i = 0; i < 4; ++i) {
            int lin = tid + i * 128;
            int r = lin >> 3, c4 = lin & 7;
            *(h4*)(&sA[buf][r * 40 + c4 * 4]) = __builtin_convertvector(ra[i], h4);
        }
        #pragma unroll
        for (int j = 0; j < 4; ++j)
            sBt[buf][(cB4 * 4 + j) * 40 + rB] = (_Float16)rb[j];
    };

    v8f acc = {};

    loadTiles(0);
    storeTiles(0);
    __syncthreads();

    const int nsteps = K >> 5;
    for (int s = 0; s < nsteps; ++s) {
        const int cur = s & 1;
        if (s + 1 < nsteps) loadTiles((s + 1) << 5);     // in-flight during WMMA

        // A fragment: lane holds row (lane&15); K = {kb..kb+7, 16+kb..16+kb+7}
        const int kb = hv * 8;
        const _Float16* pa = &sA[cur][(wave * 16 + row) * 40];
        h8 a_lo = *(const h8*)(pa + kb);
        h8 a_hi = *(const h8*)(pa + 16 + kb);
        v16h a = __builtin_shufflevector(a_lo, a_hi,
                 0,1,2,3,4,5,6,7,8,9,10,11,12,13,14,15);

        // B fragment: lane holds col (lane&15); K = {kb..kb+7, 16+kb..16+kb+7}
        const _Float16* pb = &sBt[cur][row * 40 + kb];
        h8 b_lo = *(const h8*)(pb);
        h8 b_hi = *(const h8*)(pb + 16);
        v16h b = __builtin_shufflevector(b_lo, b_hi,
                 0,1,2,3,4,5,6,7,8,9,10,11,12,13,14,15);

        acc = __builtin_amdgcn_wmma_f32_16x16x32_f16(
                  false, a, false, b, (short)0, acc, false, false);

        if (s + 1 < nsteps) storeTiles(1 - cur);         // disjoint buffer
        __syncthreads();                                 // one barrier per step
    }

    // ---- epilogue: VGPR i -> row (i + hv*8), col = lane&15
    #pragma unroll
    for (int i = 0; i < 8; ++i) {
        int r  = wave * 16 + i + hv * 8;
        int gm = m0 + r;
        if (gm < M) {
            int gn = n0 + row;
            float v = acc[i];
            if (bias) v += bias[gn];
            if (res)  v += res[(size_t)gm * N + gn];
            if (act == 1)      v = fmaxf(v, 0.0f);
            else if (act == 2) v = (v > 0.0f) ? v : (__expf(v) - 1.0f);
            C[(size_t)gm * N + gn] = v;
        }
    }
}

// ---------------------------------------------------------------------------
// BatchNorm (batch stats over all M rows), reference-style population var.
// ---------------------------------------------------------------------------
__global__ void bn_stats_kernel(const float* __restrict__ X, int M, int D,
                                float* __restrict__ mean, float* __restrict__ var)
{
    __shared__ float s1[256], s2[256];
    int d = blockIdx.x;
    float a = 0.0f, b = 0.0f;
    for (int r = threadIdx.x; r < M; r += 256) {
        float v = X[(size_t)r * D + d];
        a += v; b += v * v;
    }
    s1[threadIdx.x] = a; s2[threadIdx.x] = b;
    __syncthreads();
    for (int s = 128; s > 0; s >>= 1) {
        if (threadIdx.x < s) {
            s1[threadIdx.x] += s1[threadIdx.x + s];
            s2[threadIdx.x] += s2[threadIdx.x + s];
        }
        __syncthreads();
    }
    if (threadIdx.x == 0) {
        float m = s1[0] / (float)M;
        mean[d] = m;
        var[d]  = s2[0] / (float)M - m * m;
    }
}

__global__ void bn_apply_kernel(float* __restrict__ X, int M, int D,
                                const float* __restrict__ mean, const float* __restrict__ var,
                                const float* __restrict__ g, const float* __restrict__ b, int act)
{
    int i = blockIdx.x * blockDim.x + threadIdx.x;
    if (i >= M * D) return;
    int d = i % D;
    float v = (X[i] - mean[d]) * rsqrtf(var[d] + EPSBN) * g[d] + b[d];
    if (act == 1) v = fmaxf(v, 0.0f);
    X[i] = v;
}

// ---------------------------------------------------------------------------
// Neighborhood attention: one thread per (batch*node, head).
// q,k,v laid out (B*n, D) with head-major sub-blocks of size d = D/H.
// ---------------------------------------------------------------------------
__global__ void attn_kernel(const float* __restrict__ q, const float* __restrict__ k,
                            const float* __restrict__ v, const int* __restrict__ nbr,
                            float* __restrict__ out, int Bc, int n, int D, int kn)
{
    int t = blockIdx.x * blockDim.x + threadIdx.x;
    if (t >= Bc * n * HEADS) return;
    int hh = t % HEADS;
    int bn = t / HEADS;          // b*n + i
    int b  = bn / n;
    int d  = D / HEADS;
    float scale = rsqrtf((float)d);

    const float* qp = q + (size_t)bn * D + hh * d;
    const int*   nb = nbr + (size_t)bn * kn;

    float sc[16];
    int   id[16];
    float mx = -1e30f;
    for (int j = 0; j < kn; ++j) {
        int jj = nb[j];
        id[j] = jj;
        const float* kp = k + (size_t)(b * n + jj) * D + hh * d;
        float s = 0.0f;
        for (int x = 0; x < d; ++x) s += qp[x] * kp[x];
        s *= scale;
        sc[j] = s;
        mx = fmaxf(mx, s);
    }
    float sum = 0.0f;
    for (int j = 0; j < kn; ++j) { sc[j] = __expf(sc[j] - mx); sum += sc[j]; }
    float inv = 1.0f / sum;

    float* op = out + (size_t)bn * D + hh * d;
    for (int x = 0; x < d; ++x) {
        float accv = 0.0f;
        for (int j = 0; j < kn; ++j)
            accv += sc[j] * v[(size_t)(b * n + id[j]) * D + hh * d + x];
        op[x] = accv * inv;
    }
}

// ---------------------------------------------------------------------------
// Brute-force kNN (excl. self): keep sorted top-(kn+1), emit indices 1..kn.
// ---------------------------------------------------------------------------
__global__ void knn_kernel(const float* __restrict__ c, int Bc, int n, int kn,
                           int* __restrict__ out)
{
    int t = blockIdx.x * blockDim.x + threadIdx.x;
    if (t >= Bc * n) return;
    int b = t / n, i = t % n;
    const float* cb = c + (size_t)b * n * 3;
    float xi = cb[i * 3], yi = cb[i * 3 + 1], zi = cb[i * 3 + 2];
    const int L = kn + 1;        // <= 17
    float bd[17]; int bi[17];
    for (int j = 0; j < L; ++j) { bd[j] = 3.4e38f; bi[j] = 0; }
    for (int j = 0; j < n; ++j) {
        float dx = cb[j * 3] - xi, dy = cb[j * 3 + 1] - yi, dz = cb[j * 3 + 2] - zi;
        float d2 = dx * dx + dy * dy + dz * dz;
        if (d2 < bd[L - 1]) {
            int p = L - 1;
            while (p > 0 && bd[p - 1] > d2) { bd[p] = bd[p - 1]; bi[p] = bi[p - 1]; --p; }
            bd[p] = d2; bi[p] = j;
        }
    }
    int* op = out + (size_t)t * kn;
    for (int j = 0; j < kn; ++j) op[j] = bi[j + 1];
}

// For each fine point: nearest coarse center (argmin, first-min ties).
__global__ void assign_kernel(const float* __restrict__ c, const float* __restrict__ cn,
                              int Bc, int n, int m, int* __restrict__ out)
{
    int t = blockIdx.x * blockDim.x + threadIdx.x;
    if (t >= Bc * n) return;
    int b = t / n;
    const float* p = c + (size_t)t * 3;
    float x = p[0], y = p[1], z = p[2];
    const float* cb = cn + (size_t)b * m * 3;
    float best = 3.4e38f; int bidx = 0;
    for (int j = 0; j < m; ++j) {
        float dx = cb[j * 3] - x, dy = cb[j * 3 + 1] - y, dz = cb[j * 3 + 2] - z;
        float d2 = dx * dx + dy * dy + dz * dz;
        if (d2 < best) { best = d2; bidx = j; }
    }
    out[t] = bidx;
}

// out[b, i, :] = in[b, 4*i, :]   (m = n/4 rows out)
__global__ void stride4_gather_kernel(const float* __restrict__ in, float* __restrict__ out,
                                      int Bc, int n, int m, int D)
{
    int t = blockIdx.x * blockDim.x + threadIdx.x;
    if (t >= Bc * m * D) return;
    int dd = t % D;
    int bm = t / D;
    int b = bm / m, i = bm % m;
    out[t] = in[(size_t)(b * n + i * 4) * D + dd];
}

// out[b, i, :] = in[b, idx[b, i], :]   (n fine rows out, m coarse rows in)
__global__ void idx_gather_kernel(const float* __restrict__ in, const int* __restrict__ idx,
                                  float* __restrict__ out, int Bc, int n, int m, int D)
{
    int t = blockIdx.x * blockDim.x + threadIdx.x;
    if (t >= Bc * n * D) return;
    int dd = t % D;
    int bn = t / D;
    int b = bn / n;
    int j = idx[bn];
    out[t] = in[(size_t)(b * m + j) * D + dd];
}

__global__ void add_kernel(float* __restrict__ y, const float* __restrict__ x, int n)
{
    int i = blockIdx.x * blockDim.x + threadIdx.x;
    if (i < n) y[i] += x[i];
}

// feat (M,6) @ emb_W (6,32) — K too small for WMMA, scalar.
__global__ void emb_kernel(const float* __restrict__ feat, const float* __restrict__ W,
                           float* __restrict__ out, int M)
{
    int t = blockIdx.x * blockDim.x + threadIdx.x;
    if (t >= M * 32) return;
    int o = t & 31, m = t >> 5;
    float s = 0.0f;
    #pragma unroll
    for (int kk = 0; kk < 6; ++kk) s += feat[m * 6 + kk] * W[kk * 32 + o];
    out[t] = s;
}

// x (M,64) @ cls_W (64,13) — N=13 not a WMMA multiple, scalar.
__global__ void cls_kernel(const float* __restrict__ x, const float* __restrict__ W,
                           float* __restrict__ out, int M)
{
    int t = blockIdx.x * blockDim.x + threadIdx.x;
    if (t >= M * 13) return;
    int cc = t % 13, m = t / 13;
    float s = 0.0f;
    #pragma unroll 8
    for (int kk = 0; kk < 64; ++kk) s += x[m * 64 + kk] * W[kk * 13 + cc];
    out[t] = s;
}

// ---------------------------------------------------------------------------
// Host orchestration
// ---------------------------------------------------------------------------
static inline void gemm(hipStream_t s, const float* A, const float* Bw, float* C,
                        int M, int N, int K, const float* bias, const float* res, int act)
{
    dim3 grid(N / 16, (M + 63) / 64);
    gemm_wmma_kernel<<<grid, 128, 0, s>>>(A, Bw, C, M, N, K, bias, res, act);
}

extern "C" void kernel_launch(void* const* d_in, const int* in_sizes, int n_in,
                              void* d_out, int out_size, void* d_ws, size_t ws_size,
                              hipStream_t stream)
{
    (void)in_sizes; (void)n_in; (void)out_size; (void)ws_size;

    // ---- parameter pointers, in setup_inputs() flattening order --------------
    int pi = 0;
    auto nf = [&]() -> const float* { return (const float*)d_in[pi++]; };

    const float* coord = nf();
    const float* feat  = nf();
    const float* emb_W = nf();
    const float* emb_g = nf();
    const float* emb_b = nf();

    struct GT { const float *Wq,*Wk,*Wv,*Wo,*g1,*b1,*W1,*W2,*g2,*b2; };
    auto readGT = [&]() {
        GT g;
        g.Wq = nf(); g.Wk = nf(); g.Wv = nf(); g.Wo = nf();
        g.g1 = nf(); g.b1 = nf(); g.W1 = nf(); g.W2 = nf();
        g.g2 = nf(); g.b2 = nf();
        return g;
    };
    GT mp_down[4][2];
    for (int l = 0; l < 4; ++l) for (int s = 0; s < 2; ++s) mp_down[l][s] = readGT();
    GT bott[2];
    for (int s = 0; s < 2; ++s) bott[s] = readGT();
    struct DS { const float *W, *g, *b; } down[4];
    for (int l = 0; l < 4; ++l) { down[l].W = nf(); down[l].g = nf(); down[l].b = nf(); }
    const float* upW[4];
    for (int l = 0; l < 4; ++l) upW[l] = nf();
    GT mp_up[4][2];
    for (int l = 0; l < 4; ++l) for (int s = 0; s < 2; ++s) mp_up[l][s] = readGT();
    struct PP { const float *W, *b; } post[3];
    for (int i = 0; i < 3; ++i) { post[i].W = nf(); post[i].b = nf(); }
    const float* bn2_g = nf();
    const float* bn2_b = nf();
    const float* cls_W = nf();

    // ---- workspace bump allocator -------------------------------------------
    char* wp = (char*)d_ws;
    auto falloc = [&](size_t elems) -> float* {
        float* p = (float*)wp;
        wp += ((elems * sizeof(float)) + 255) & ~((size_t)255);
        return p;
    };
    auto ialloc = [&](size_t elems) -> int* {
        int* p = (int*)wp;
        wp += ((elems * sizeof(int)) + 255) & ~((size_t)255);
        return p;
    };

    const int Bc = 2;
    const int ns[5]  = {4096, 1024, 256, 64, 16};
    const int Dsz[5] = {32, 64, 128, 256, 512};
    const size_t BUF = 8192 * 64;        // max of M*N, M*2D across the net

    float* h    = falloc(BUF);
    float* qb   = falloc(8192 * 32);
    float* kbuf = falloc(8192 * 32);
    float* vb   = falloc(8192 * 32);
    float* ob   = falloc(8192 * 32);
    float* mid  = falloc(BUF);
    float* tb   = falloc(BUF);
    float* meanb = falloc(1024);
    float* varb  = falloc(1024);

    float* cl[5];
    cl[0] = (float*)coord;               // level-0 coords are the input (read-only)
    for (int l = 1; l < 5; ++l) cl[l] = falloc((size_t)Bc * ns[l] * 3);

    float* skip[4];
    for (int l = 0; l < 4; ++l) skip[l] = falloc((size_t)Bc * ns[l] * Dsz[l]);

    int* nbr[4];
    for (int l = 0; l < 4; ++l) nbr[l] = ialloc((size_t)Bc * ns[l] * 16);
    int* nbr_bot = ialloc((size_t)Bc * 16 * 15);
    int* assigns[4];
    for (int l = 0; l < 4; ++l) assigns[l] = ialloc((size_t)Bc * ns[l]);

    auto cdiv = [](int a, int b) { return (a + b - 1) / b; };

    auto bn = [&](float* X, int M, int D, const float* g, const float* b, int act) {
        bn_stats_kernel<<<D, 256, 0, stream>>>(X, M, D, meanb, varb);
        bn_apply_kernel<<<cdiv(M * D, 256), 256, 0, stream>>>(X, M, D, meanb, varb, g, b, act);
    };

    auto gt_layer = [&](float* hc, const GT& g, const int* nb, int n, int D, int kn) {
        int M = Bc * n;
        gemm(stream, hc, g.Wq, qb,   M, D, D, nullptr, nullptr, 0);
        gemm(stream, hc, g.Wk, kbuf, M, D, D, nullptr, nullptr, 0);
        gemm(stream, hc, g.Wv, vb,   M, D, D, nullptr, nullptr, 0);
        attn_kernel<<<cdiv(M * HEADS, 256), 256, 0, stream>>>(qb, kbuf, vb, nb, ob, Bc, n, D, kn);
        gemm(stream, ob, g.Wo, tb, M, D, D, nullptr, hc, 0);           // tb = attn@Wo + h
        bn(tb, M, D, g.g1, g.b1, 0);
        gemm(stream, tb, g.W1, mid, M, 2 * D, D, nullptr, nullptr, 1); // relu
        gemm(stream, mid, g.W2, hc, M, D, 2 * D, nullptr, tb, 0);      // h = mid@W2 + tb
        bn(hc, M, D, g.g2, g.b2, 0);
    };

    // ---- embedding -----------------------------------------------------------
    {
        int M = Bc * ns[0];
        emb_kernel<<<cdiv(M * 32, 256), 256, 0, stream>>>(feat, emb_W, h, M);
        bn(h, M, 32, emb_g, emb_b, 1);
    }

    // ---- encoder -------------------------------------------------------------
    for (int l = 0; l < 4; ++l) {
        int n = ns[l], D = Dsz[l], M = Bc * n;
        knn_kernel<<<cdiv(Bc * n, 128), 128, 0, stream>>>(cl[l], Bc, n, 16, nbr[l]);
        for (int s = 0; s < 2; ++s) gt_layer(h, mp_down[l][s], nbr[l], n, D, 16);
        hipMemcpyAsync(skip[l], h, (size_t)M * D * sizeof(float),
                       hipMemcpyDeviceToDevice, stream);
        // f = relu(bn(h @ down.W)) at FULL resolution (stats before subsample)
        int Dn = Dsz[l + 1];
        gemm(stream, h, down[l].W, tb, M, Dn, D, nullptr, nullptr, 0);
        bn(tb, M, Dn, down[l].g, down[l].b, 1);
        // coords / assignments / subsample
        int m = ns[l + 1];
        stride4_gather_kernel<<<cdiv(Bc * m * 3, 256), 256, 0, stream>>>(cl[l], cl[l + 1], Bc, n, m, 3);
        assign_kernel<<<cdiv(Bc * n, 128), 128, 0, stream>>>(cl[l], cl[l + 1], Bc, n, m, assigns[l]);
        stride4_gather_kernel<<<cdiv(Bc * m * Dn, 256), 256, 0, stream>>>(tb, h, Bc, n, m, Dn);
    }

    // ---- bottleneck ----------------------------------------------------------
    knn_kernel<<<cdiv(Bc * 16, 64), 64, 0, stream>>>(cl[4], Bc, 16, 15, nbr_bot);
    for (int s = 0; s < 2; ++s) gt_layer(h, bott[s], nbr_bot, 16, 512, 15);

    // ---- decoder -------------------------------------------------------------
    for (int li = 0; li < 4; ++li) {
        int lc = 4 - li;                 // coarse level
        int lf = 3 - li;                 // fine level
        int Mc = Bc * ns[lc], Df = Dsz[lf], Dc = Dsz[lc];
        gemm(stream, h, upW[li], tb, Mc, Df, Dc, nullptr, nullptr, 0);
        idx_gather_kernel<<<cdiv(Bc * ns[lf] * Df, 256), 256, 0, stream>>>(
            tb, assigns[lf], h, Bc, ns[lf], ns[lc], Df);
        for (int s = 0; s < 2; ++s) gt_layer(h, mp_up[li][s], nbr[lf], ns[lf], Df, 16);
        add_kernel<<<cdiv(Bc * ns[lf] * Df, 256), 256, 0, stream>>>(
            h, skip[lf], Bc * ns[lf] * Df);
    }

    // ---- post MLP + final BN + classifier ------------------------------------
    {
        int M = Bc * ns[0];
        gemm(stream, h,   post[0].W, tb,  M, 64, 32, post[0].b, nullptr, 2); // elu
        gemm(stream, tb,  post[1].W, mid, M, 64, 64, post[1].b, nullptr, 2);
        gemm(stream, mid, post[2].W, tb,  M, 64, 64, post[2].b, nullptr, 2);
        bn(tb, M, 64, bn2_g, bn2_b, 0);
        cls_kernel<<<cdiv(M * 13, 256), 256, 0, stream>>>(tb, cls_W, (float*)d_out, M);
    }
}
